// MVMP_5111011083054
// MI455X (gfx1250) — compile-verified, matrix operand order checked by reading, not run on hardware
//
#include <hip/hip_runtime.h>

// ---------------------------------------------------------------------------
// MI455X (gfx1250) fused graph-transformer.
// All dense projections run through v_wmma_f32_16x16x32_bf16 (wave32 WMMA).
// Weights are pre-converted to bf16 once per launch; activations convert
// during LDS staging (packed b128/b32 ds stores).
// ---------------------------------------------------------------------------

typedef __bf16 bf16_t;
typedef __attribute__((ext_vector_type(16))) __bf16 v16bf;
typedef __attribute__((ext_vector_type(8)))  __bf16 v8bf;
typedef __attribute__((ext_vector_type(4)))  __bf16 v4bf;
typedef __attribute__((ext_vector_type(2)))  __bf16 v2bf;
typedef __attribute__((ext_vector_type(8)))  float  v8f;

#define GN 50000
#define GD 300
#define GDEG 6
#define GH 4
#define GDK 75
#define GE (GN * GDEG)

// A-row mapping modes (fused gathers), compile-time
#define MODE_IDENT  0
#define MODE_GATHER 1   // row = idx[m]
#define MODE_DIV6   2   // row = m / 6   (broadcast node feature over mailbox)
#define MODE_XOR1   3   // row = m ^ 1   (reverse-edge permutation)
#define MODE_NONE   4

// epilogue flags, compile-time
#define EPI_RELU 1
#define EPI_BETA 2      // accumulate onto existing C
#define EPI_BIAS 4
#define EPI_RES  8

__device__ __forceinline__ bf16_t to_bf16(float f) {
    union { float f; unsigned u; } a; a.f = f;
    unsigned r = a.u + 0x7FFFu + ((a.u >> 16) & 1u);   // round-to-nearest-even
    unsigned short h = (unsigned short)(r >> 16);
    union { unsigned short s; bf16_t b; } c; c.s = h;
    return c.b;
}

template <int MODE>
__device__ __forceinline__ long map_row(const int* __restrict__ idx, long m) {
    if constexpr (MODE == MODE_IDENT)  return m;
    if constexpr (MODE == MODE_GATHER) return (long)idx[m];
    if constexpr (MODE == MODE_DIV6)   return m / 6;
    return m ^ 1;   // MODE_XOR1
}

// ---------------------------------------------------------------------------
// weight pre-conversion: f32 -> bf16 (once per launch)
// ---------------------------------------------------------------------------
__global__ void f32_to_bf16_kernel(const float* __restrict__ in,
                                   bf16_t* __restrict__ out, int n)
{
    int t = blockIdx.x * blockDim.x + threadIdx.x;
    if (t < n) out[t] = to_bf16(in[t]);
}

// ---------------------------------------------------------------------------
// Generic bf16 WMMA GEMM:
//   C[M,Nn] = epi( (A0[map0(m)] + sign1*A1[map1(m)]) @ B[K,Nn] )
// 128 threads = 4 waves; 64x64 block tile; each wave a 32x32 quadrant
// (2x2 wmma tiles, fragments reused). A staged to LDS with packed b128
// stores; bf16 B staged transposed with packed-pair b32 stores so fragment
// reads are ds_load_b128.
// ---------------------------------------------------------------------------
template <int MODE0, int MODE1, int FLAGS>
__launch_bounds__(128)
__global__ void gemm_wmma(
    const float* __restrict__ A0, const int* __restrict__ idx0,
    const float* __restrict__ A1, float sign1,
    const bf16_t* __restrict__ B, const float* __restrict__ bias,
    const float* __restrict__ residual, float* __restrict__ C,
    int M, int K, int Nn)
{
    constexpr int BM = 64, BN = 64, LDA = 40;   // LDA*2B = 80B rows (16B aligned)
    __shared__ __align__(16) bf16_t lAs[BM][LDA];
    __shared__ __align__(16) bf16_t lBt[BN][LDA];   // transposed: [col][k]

    const int tid  = threadIdx.x;
    const int lane = tid & 31;
    const int wave = tid >> 5;
    const int bm   = blockIdx.y * BM;
    const int bn   = blockIdx.x * BN;
    const int wm   = (wave >> 1) * 32;
    const int wn   = (wave & 1) * 32;

    v8f acc[2][2] = {};

    for (int k0 = 0; k0 < K; k0 += 32) {
        const bool fullK = (k0 + 32 <= K);

        if (k0 + 32 < K)
            __builtin_prefetch(&B[(long)(k0 + 32) * Nn + bn + lane], 0, 1);

        // ---- A tile: 64 rows x 32 cols; 2 threads per row, 16 cols each ----
        {
            const int row = tid >> 1;
            const int cb  = (tid & 1) * 16;
            const long m  = (long)bm + row;
            float va[16];
            if (m < M) {
                const long r0 = map_row<MODE0>(idx0, m);
                const float* p0 = A0 + r0 * (long)K + k0 + cb;
                const float* p1 = nullptr;
                if constexpr (MODE1 != MODE_NONE) {
                    const long r1 = map_row<MODE1>(idx0, m);
                    p1 = A1 + r1 * (long)K + k0 + cb;
                }
                if (fullK) {
                    const float4* q0 = (const float4*)p0;
                    #pragma unroll
                    for (int j = 0; j < 4; ++j) {
                        float4 t = q0[j];
                        va[4*j+0] = t.x; va[4*j+1] = t.y;
                        va[4*j+2] = t.z; va[4*j+3] = t.w;
                    }
                    if constexpr (MODE1 != MODE_NONE) {
                        const float4* q1 = (const float4*)p1;
                        #pragma unroll
                        for (int j = 0; j < 4; ++j) {
                            float4 t = q1[j];
                            va[4*j+0] += sign1 * t.x; va[4*j+1] += sign1 * t.y;
                            va[4*j+2] += sign1 * t.z; va[4*j+3] += sign1 * t.w;
                        }
                    }
                } else {
                    #pragma unroll
                    for (int j = 0; j < 16; ++j) {
                        float v = 0.f;
                        if (k0 + cb + j < K) {
                            v = p0[j];
                            if constexpr (MODE1 != MODE_NONE) v += sign1 * p1[j];
                        }
                        va[j] = v;
                    }
                }
            } else {
                #pragma unroll
                for (int j = 0; j < 16; ++j) va[j] = 0.f;
            }
            v8bf pa0, pa1;
            #pragma unroll
            for (int j = 0; j < 8; ++j) {
                pa0[j] = to_bf16(va[j]);
                pa1[j] = to_bf16(va[8 + j]);
            }
            *(v8bf*)&lAs[row][cb]     = pa0;   // ds_store_b128
            *(v8bf*)&lAs[row][cb + 8] = pa1;   // ds_store_b128
        }

        // ---- B tile (bf16): 2 k-rows x 8 cols per thread; transposed store
        //      as packed K-pairs -> 8x ds_store_b32 per thread ----
        {
            const int kr = (tid >> 3) * 2;        // even k within tile
            const int cb = (tid & 7) * 8;         // col base
            bf16_t b0[8], b1[8];
            if (fullK && bn + cb + 8 <= Nn) {
                const v4bf* q0 = (const v4bf*)&B[(long)(k0 + kr) * Nn + bn + cb];
                const v4bf* q1 = (const v4bf*)&B[(long)(k0 + kr + 1) * Nn + bn + cb];
                v4bf a0 = q0[0], a1 = q0[1];
                v4bf c0 = q1[0], c1 = q1[1];
                #pragma unroll
                for (int j = 0; j < 4; ++j) {
                    b0[j] = a0[j]; b0[4 + j] = a1[j];
                    b1[j] = c0[j]; b1[4 + j] = c1[j];
                }
            } else {
                #pragma unroll
                for (int j = 0; j < 8; ++j) {
                    int n = bn + cb + j;
                    bool nok = (n < Nn);
                    b0[j] = (nok && k0 + kr     < K) ? B[(long)(k0 + kr)     * Nn + n]
                                                     : (bf16_t)0.f;
                    b1[j] = (nok && k0 + kr + 1 < K) ? B[(long)(k0 + kr + 1) * Nn + n]
                                                     : (bf16_t)0.f;
                }
            }
            #pragma unroll
            for (int j = 0; j < 8; ++j) {
                v2bf p; p[0] = b0[j]; p[1] = b1[j];
                *(v2bf*)&lBt[cb + j][kr] = p;      // ds_store_b32
            }
        }

        __syncthreads();

        // ---- fragments (ds_load_b128 pairs) + 2x2 wmma ----
        // ISA 16-bit 16x32 A layout:
        //   lanes 0-15 : row=lane,    bf[0..7]=K0..7,  bf[8..15]=K16..23
        //   lanes16-31 : row=lane-16, bf[0..7]=K8..15, bf[8..15]=K24..31
        const int kb = (lane >> 4) * 8;
        v16bf af[2], bfv[2];
        #pragma unroll
        for (int mi = 0; mi < 2; ++mi) {
            const int arow = wm + mi * 16 + (lane & 15);
            v8bf lo = *(const v8bf*)&lAs[arow][kb];
            v8bf hi = *(const v8bf*)&lAs[arow][16 + kb];
            af[mi] = __builtin_shufflevector(lo, hi,
                     0,1,2,3,4,5,6,7,8,9,10,11,12,13,14,15);
        }
        #pragma unroll
        for (int ni = 0; ni < 2; ++ni) {
            const int bcol = wn + ni * 16 + (lane & 15);
            v8bf lo = *(const v8bf*)&lBt[bcol][kb];
            v8bf hi = *(const v8bf*)&lBt[bcol][16 + kb];
            bfv[ni] = __builtin_shufflevector(lo, hi,
                      0,1,2,3,4,5,6,7,8,9,10,11,12,13,14,15);
        }
        #pragma unroll
        for (int mi = 0; mi < 2; ++mi)
            #pragma unroll
            for (int ni = 0; ni < 2; ++ni)
                acc[mi][ni] = __builtin_amdgcn_wmma_f32_16x16x32_bf16(
                    false, af[mi], false, bfv[ni], (short)0, acc[mi][ni],
                    false, false);

        __syncthreads();
    }

    // ---- epilogue: lane col = lane&15, rows = 8*(lane>>4)+r ----
    #pragma unroll
    for (int mi = 0; mi < 2; ++mi) {
        #pragma unroll
        for (int ni = 0; ni < 2; ++ni) {
            const int col   = bn + wn + ni * 16 + (lane & 15);
            const int rbase = bm + wm + mi * 16 + 8 * (lane >> 4);
            if (col < Nn) {
                #pragma unroll
                for (int r = 0; r < 8; ++r) {
                    const int row = rbase + r;
                    if (row < M) {
                        float v = acc[mi][ni][r];
                        if constexpr (FLAGS & EPI_BIAS) v += bias[col];
                        if constexpr (FLAGS & EPI_RES)  v += residual[(long)row * Nn + col];
                        if constexpr (FLAGS & EPI_BETA) v += C[(long)row * Nn + col];
                        if constexpr (FLAGS & EPI_RELU) v = fmaxf(v, 0.f);
                        C[(long)row * Nn + col] = v;
                    }
                }
            }
        }
    }
}

// ---------------------------------------------------------------------------
// attention scores + softmax : one thread per (node, head)
// ---------------------------------------------------------------------------
__global__ void attn_scores(const float* __restrict__ q,
                            const float* __restrict__ kk,
                            float* __restrict__ attn)
{
    int t = blockIdx.x * blockDim.x + threadIdx.x;
    if (t >= GN * GH) return;
    int n = t >> 2, hh = t & 3;
    const float scale = 0.115470053838f;            // 1/sqrt(75)
    const float* qp = q + (long)n * GD + hh * GDK;
    float s[GDEG];
    float mx = -1e30f;
    #pragma unroll
    for (int d = 0; d < GDEG; ++d) {
        const float* kp = kk + ((long)n * GDEG + d) * GD + hh * GDK;
        float a = 0.f;
        for (int c = 0; c < GDK; ++c) a += qp[c] * kp[c];
        s[d] = a * scale;
        mx = fmaxf(mx, s[d]);
    }
    float sum = 0.f;
    #pragma unroll
    for (int d = 0; d < GDEG; ++d) { s[d] = __expf(s[d] - mx); sum += s[d]; }
    float inv = 1.f / sum;
    float* ap = attn + (long)t * GDEG;
    #pragma unroll
    for (int d = 0; d < GDEG; ++d) ap[d] = s[d] * inv;
}

// ---------------------------------------------------------------------------
// attention combine : one thread per (node, channel)
// ---------------------------------------------------------------------------
__global__ void attn_combine(const float* __restrict__ attn,
                             const float* __restrict__ vv,
                             float* __restrict__ out)
{
    long t = (long)blockIdx.x * blockDim.x + threadIdx.x;
    if (t >= (long)GN * GD) return;
    int n = (int)(t / GD), c = (int)(t % GD);
    int hh = c / GDK;
    const float* ap = attn + ((long)n * GH + hh) * GDEG;
    float a = 0.f;
    #pragma unroll
    for (int d = 0; d < GDEG; ++d)
        a += ap[d] * vv[((long)n * GDEG + d) * GD + c];
    out[t] = a;
}

// ---------------------------------------------------------------------------
// mailbox segment-sum: every node has exactly 6 in-edges (nbr_eids)
// ---------------------------------------------------------------------------
__global__ void mail_sum(const float* __restrict__ h,
                         const int* __restrict__ nbr,
                         float* __restrict__ mail)
{
    long t = (long)blockIdx.x * blockDim.x + threadIdx.x;
    if (t >= (long)GN * GD) return;
    int n = (int)(t / GD), c = (int)(t % GD);
    float a = 0.f;
    #pragma unroll
    for (int d = 0; d < GDEG; ++d)
        a += h[(long)nbr[n * GDEG + d] * GD + c];
    mail[t] = a;
}

// ---------------------------------------------------------------------------
extern "C" void kernel_launch(void* const* d_in, const int* in_sizes, int n_in,
                              void* d_out, int out_size, void* d_ws, size_t ws_size,
                              hipStream_t stream)
{
    (void)in_sizes; (void)n_in; (void)out_size; (void)ws_size;

    const float* f      = (const float*)d_in[0];
    const float* x      = (const float*)d_in[1];
    const float* Wq     = (const float*)d_in[2];
    const float* bq     = (const float*)d_in[3];
    const float* Wk     = (const float*)d_in[4];
    const float* bk     = (const float*)d_in[5];
    const float* Wv     = (const float*)d_in[6];
    const float* bv     = (const float*)d_in[7];
    const float* Wo     = (const float*)d_in[8];
    const float* bo     = (const float*)d_in[9];
    const float* W_mp   = (const float*)d_in[10];
    const float* b_mp   = (const float*)d_in[11];
    const float* W_last = (const float*)d_in[12];
    const float* b_last = (const float*)d_in[13];
    const int*   src    = (const int*)d_in[14];
    const int*   nbr    = (const int*)d_in[16];

    const size_t NF = (size_t)GN * GD;    // 15.0M floats
    const size_t EF = (size_t)GE * GD;    // 90.0M floats
    const int    WW = GD * GD;            // 90000 (one weight matrix)

    float* ws   = (float*)d_ws;
    float* f_h  = ws;                       // [N,D]
    float* qb   = f_h + NF;                 // [N,D]  (q, then reused as att_out)
    float* kkb  = qb + NF;                  // [E,D]
    float* vvb  = kkb + EF;                 // [E,D]
    float* hA   = vvb + EF;                 // [E,D]  layer-0 edge state
    float* attn = hA + EF;                  // [N,H,DEG]
    float* mail = attn + (size_t)GN * GH * GDEG;  // [N,D]
    bf16_t* wbf = (bf16_t*)(mail + NF);     // bf16 weights: 9 blocks of [D,D]

    bf16_t* bWq = wbf;                      // block offsets into wbf
    bf16_t* bWk = wbf + 1 * (size_t)WW;
    bf16_t* bWv = wbf + 2 * (size_t)WW;
    bf16_t* bWo = wbf + 3 * (size_t)WW;
    bf16_t* bWm0 = wbf + 4 * (size_t)WW;
    bf16_t* bWm1 = wbf + 5 * (size_t)WW;
    bf16_t* bWl  = wbf + 6 * (size_t)WW;    // 3 consecutive blocks [3D,D]

    float* fout = (float*)d_out;            // f_final [N,D]
    float* hout = fout + NF;                // h       [E,D]

    const dim3 blk(128);
    const dim3 gridN((GD + 63) / 64, (GN + 63) / 64);
    const dim3 gridE((GD + 63) / 64, (GE + 63) / 64);
    const int T = 256;

    // ---- pre-convert all weights to bf16 (deterministic, every call) ----
    const int cvtBlk = (WW + T - 1) / T;
    f32_to_bf16_kernel<<<cvtBlk, T, 0, stream>>>(Wq, bWq, WW);
    f32_to_bf16_kernel<<<cvtBlk, T, 0, stream>>>(Wk, bWk, WW);
    f32_to_bf16_kernel<<<cvtBlk, T, 0, stream>>>(Wv, bWv, WW);
    f32_to_bf16_kernel<<<cvtBlk, T, 0, stream>>>(Wo, bWo, WW);
    f32_to_bf16_kernel<<<cvtBlk, T, 0, stream>>>(W_mp, bWm0, WW);
    f32_to_bf16_kernel<<<cvtBlk, T, 0, stream>>>(W_mp + (size_t)WW, bWm1, WW);
    f32_to_bf16_kernel<<<(3 * WW + T - 1) / T, T, 0, stream>>>(W_last, bWl, 3 * WW);

    const float* h_in = x;
    const float* f_in = f;
    for (int layer = 0; layer < 2; ++layer) {
        // q = f_h @ Wq + bq
        gemm_wmma<MODE_IDENT, MODE_NONE, EPI_BIAS><<<gridN, blk, 0, stream>>>(
            f_in, nullptr, nullptr, 0.f, bWq, bq, nullptr, qb, GN, GD, GD);
        // k = h[nbr] @ Wk + bk        (gather fused)
        gemm_wmma<MODE_GATHER, MODE_NONE, EPI_BIAS><<<gridE, blk, 0, stream>>>(
            h_in, nbr, nullptr, 0.f, bWk, bk, nullptr, kkb, GE, GD, GD);
        // v = (h[nbr] + f_h[node]) @ Wv + bv   (gather + broadcast fused)
        gemm_wmma<MODE_GATHER, MODE_DIV6, EPI_BIAS><<<gridE, blk, 0, stream>>>(
            h_in, nbr, f_in, 1.f, bWv, bv, nullptr, vvb, GE, GD, GD);
        // softmax attention over DEG=6
        attn_scores<<<(GN * GH + T - 1) / T, T, 0, stream>>>(qb, kkb, attn);
        attn_combine<<<(int)((NF + T - 1) / T), T, 0, stream>>>(attn, vvb, qb);
        // f_h = att_out @ Wo + bo
        gemm_wmma<MODE_IDENT, MODE_NONE, EPI_BIAS><<<gridN, blk, 0, stream>>>(
            qb, nullptr, nullptr, 0.f, bWo, bo, nullptr, f_h, GN, GD, GD);
        // h = relu(x + (f_h[src] - h[e^1]) @ W_mp[layer] + b_mp[layer])
        float* h_out = (layer == 0) ? hA : hout;   // last layer -> output region
        gemm_wmma<MODE_GATHER, MODE_XOR1, EPI_BIAS | EPI_RES | EPI_RELU>
            <<<gridE, blk, 0, stream>>>(
            f_h, src, h_in, -1.f,
            (layer == 0) ? bWm0 : bWm1, b_mp + (size_t)layer * GD,
            x, h_out, GE, GD, GD);
        h_in = h_out;
        f_in = f_h;
    }

    // mail = segment_sum(h, dst)  (each node: its 6 in-edge ids)
    mail_sum<<<(int)((NF + T - 1) / T), T, 0, stream>>>(hout, nbr, mail);

    // f_final = [mail | f_h | f] @ W_last + b_last  (three accumulating GEMMs)
    gemm_wmma<MODE_IDENT, MODE_NONE, EPI_BIAS><<<gridN, blk, 0, stream>>>(
        mail, nullptr, nullptr, 0.f, bWl, b_last, nullptr, fout, GN, GD, GD);
    gemm_wmma<MODE_IDENT, MODE_NONE, EPI_BETA><<<gridN, blk, 0, stream>>>(
        f_h, nullptr, nullptr, 0.f, bWl + (size_t)WW, nullptr, nullptr,
        fout, GN, GD, GD);
    gemm_wmma<MODE_IDENT, MODE_NONE, EPI_BETA><<<gridN, blk, 0, stream>>>(
        f, nullptr, nullptr, 0.f, bWl + 2 * (size_t)WW, nullptr, nullptr,
        fout, GN, GD, GD);
}
